// SwinUNet_85916525789555
// MI455X (gfx1250) — matrix operand order repaired and checked by reading, not compiled, not run on hardware
//
#include <hip/hip_runtime.h>
#include <hip/hip_bf16.h>
#include <math.h>

typedef __attribute__((ext_vector_type(16))) _Float16 v16h;
typedef __attribute__((ext_vector_type(8)))  _Float16 v8h;
typedef __attribute__((ext_vector_type(8)))  float    v8f;

#define LOG100 4.6051701859880913680f

__device__ inline float wave_sum(float v) {
#pragma unroll
  for (int off = 16; off > 0; off >>= 1) v += __shfl_xor(v, off, 32);
  return v;
}

// ---------------- f32 -> f16 flat convert (weights) ----------------
__global__ void cvt_f16_kernel(const float* __restrict__ in, _Float16* __restrict__ out, int n) {
  int i = blockIdx.x * blockDim.x + threadIdx.x;
  if (i < n) out[i] = (_Float16)in[i];
}

// ------------- qkv bias with zeroed K slice (SwinV2) ---------------
__global__ void prep_bqkv_kernel(const float* __restrict__ b, float* __restrict__ out, int C) {
  int i = blockIdx.x * blockDim.x + threadIdx.x;
  if (i < 3 * C) out[i] = (i >= C && i < 2 * C) ? 0.f : b[i];
}

// --- continuous position bias: 16*sigmoid(MLP(table))[rel] -> (4,16,16) ---
__global__ __launch_bounds__(256) void cpb_kernel(const float* __restrict__ w1,
                                                  const float* __restrict__ b1,
                                                  const float* __restrict__ w2,
                                                  float* __restrict__ out) {
  __shared__ float tab[49 * 4];
  int t = threadIdx.x;
  if (t < 196) {
    int idx = t >> 2, head = t & 3;
    float dh = (float)(idx / 7 - 3), dw = (float)(idx % 7 - 3);
    float vh = dh * (8.f / 3.f), vw = dw * (8.f / 3.f);
    float t0 = copysignf(log2f(fabsf(vh) + 1.f) * (1.f / 3.f), vh);
    float t1 = copysignf(log2f(fabsf(vw) + 1.f) * (1.f / 3.f), vw);
    float acc = 0.f;
    for (int j = 0; j < 512; ++j) {
      float h = fmaxf(0.f, t0 * w1[2 * j] + t1 * w1[2 * j + 1] + b1[j]);
      acc += h * w2[head * 512 + j];
    }
    tab[idx * 4 + head] = acc;
  }
  __syncthreads();
  for (int o = t; o < 1024; o += 256) {
    int head = o >> 8, p = o & 255;
    int i = p >> 4, j = p & 15;
    int rel = (i / 4 - j / 4 + 3) * 7 + (i % 4 - j % 4 + 3);
    out[o] = 16.f / (1.f + expf(-tab[rel * 4 + head]));
  }
}

// ---- patch embed: 1x1 conv 3->64 + LN(64); one wave32 per token ----
__global__ __launch_bounds__(256) void patch_embed_kernel(
    const float* __restrict__ x, const float* __restrict__ cw, const float* __restrict__ cb,
    const float* __restrict__ g, const float* __restrict__ bt,
    float* __restrict__ p32, _Float16* __restrict__ p16, int Bn, int H, int W) {
  int wave = (blockIdx.x * blockDim.x + threadIdx.x) >> 5;
  int lane = threadIdx.x & 31;
  int T = Bn * H * W;
  if (wave >= T) return;
  int b = wave / (H * W);
  int rem = wave % (H * W);
  size_t plane = (size_t)H * W;
  float x0 = x[((size_t)b * 3 + 0) * plane + rem];
  float x1 = x[((size_t)b * 3 + 1) * plane + rem];
  float x2 = x[((size_t)b * 3 + 2) * plane + rem];
  float v[2]; float s = 0.f, s2 = 0.f;
#pragma unroll
  for (int e = 0; e < 2; ++e) {
    int c = lane + 32 * e;
    float y = x0 * cw[c] + x1 * cw[64 + c] + x2 * cw[128 + c] + cb[c];
    v[e] = y; s += y; s2 += y * y;
  }
  s = wave_sum(s); s2 = wave_sum(s2);
  float mu = s * (1.f / 64.f);
  float inv = rsqrtf(s2 * (1.f / 64.f) - mu * mu + 1e-5f);
#pragma unroll
  for (int e = 0; e < 2; ++e) {
    int c = lane + 32 * e;
    float y = (v[e] - mu) * inv * g[c] + bt[c];
    size_t o = (size_t)wave * 64 + c;
    p32[o] = y; p16[o] = (_Float16)y;
  }
}

// ------------- LayerNorm (+optional residual) per token wave -------------
template <int C>
__global__ __launch_bounds__(256) void ln_kernel(
    const float* __restrict__ in, const float* __restrict__ resid,
    const float* __restrict__ g, const float* __restrict__ bt,
    float* __restrict__ out32, _Float16* __restrict__ out16, int T) {
  constexpr int NE = C / 32;
  int wave = (blockIdx.x * blockDim.x + threadIdx.x) >> 5;
  int lane = threadIdx.x & 31;
  if (wave >= T) return;
  const float* r = in + (size_t)wave * C;
  float v[NE]; float s = 0.f, s2 = 0.f;
#pragma unroll
  for (int e = 0; e < NE; ++e) {
    float y = r[lane + 32 * e];
    v[e] = y; s += y; s2 += y * y;
  }
  s = wave_sum(s); s2 = wave_sum(s2);
  float mu = s * (1.f / C);
  float inv = rsqrtf(s2 * (1.f / C) - mu * mu + 1e-5f);
#pragma unroll
  for (int e = 0; e < NE; ++e) {
    int c = lane + 32 * e;
    float y = (v[e] - mu) * inv * g[c] + bt[c];
    if (resid) y += resid[(size_t)wave * C + c];
    size_t o = (size_t)wave * C + c;
    if (out32) out32[o] = y;
    if (out16) out16[o] = (_Float16)y;
  }
}

// --------- WMMA GEMM: out[M,N] = A[M,K](f16) @ Bw[N,K]^T + bias ----------
// one wave -> 32x64 output block (2 M-tiles x 4 N-tiles, register-blocked).
// All 6 fragments of a K-step are loaded into distinct registers before the
// 8 WMMAs issue, so loadcnt waits stagger instead of serializing per-WMMA.
union Frag16 { v16h h; v8h q[2]; };

template <bool GELU, bool OUT32, bool OUT16>
__global__ __launch_bounds__(256) void gemm_wmma_kernel(
    const _Float16* __restrict__ A, const _Float16* __restrict__ Bw,
    const float* __restrict__ bias,
    float* __restrict__ out32, _Float16* __restrict__ out16,
    int M, int N, int K) {
  int lane = threadIdx.x & 31;
  int gw = blockIdx.x * (blockDim.x >> 5) + (threadIdx.x >> 5);
  int nChunks = N >> 6;
  int mBlocks = M >> 5;
  if (gw >= mBlocks * nChunks) return;      // wave-uniform: EXEC stays all-ones
  int mB = gw / nChunks, nC = gw % nChunks;
  const _Float16* aBase0 = A + (size_t)(mB * 32 + (lane & 15)) * K + ((lane < 16) ? 0 : 8);
  const _Float16* aBase1 = aBase0 + (size_t)16 * K;
  int kbb = (lane < 16) ? 0 : 8;
  const _Float16* bp0 = Bw + (size_t)(nC * 64 + (lane & 15)) * K + kbb;
  const _Float16* bp1 = bp0 + (size_t)16 * K;
  const _Float16* bp2 = bp0 + (size_t)32 * K;
  const _Float16* bp3 = bp0 + (size_t)48 * K;
  v8f acc[2][4];
#pragma unroll
  for (int m = 0; m < 2; ++m)
#pragma unroll
    for (int n = 0; n < 4; ++n)
#pragma unroll
      for (int r = 0; r < 8; ++r) acc[m][n][r] = 0.f;
#pragma unroll 2
  for (int k0 = 0; k0 < K; k0 += 32) {
    Frag16 a0, a1, f0, f1, f2, f3;
    a0.q[0] = *(const v8h*)(aBase0 + k0); a0.q[1] = *(const v8h*)(aBase0 + k0 + 16);
    a1.q[0] = *(const v8h*)(aBase1 + k0); a1.q[1] = *(const v8h*)(aBase1 + k0 + 16);
    f0.q[0] = *(const v8h*)(bp0 + k0);    f0.q[1] = *(const v8h*)(bp0 + k0 + 16);
    f1.q[0] = *(const v8h*)(bp1 + k0);    f1.q[1] = *(const v8h*)(bp1 + k0 + 16);
    f2.q[0] = *(const v8h*)(bp2 + k0);    f2.q[1] = *(const v8h*)(bp2 + k0 + 16);
    f3.q[0] = *(const v8h*)(bp3 + k0);    f3.q[1] = *(const v8h*)(bp3 + k0 + 16);
    if (k0 + 32 < K) {
      __builtin_prefetch((const void*)(aBase0 + k0 + 32), 0, 1);
      __builtin_prefetch((const void*)(aBase1 + k0 + 32), 0, 1);
    }
    acc[0][0] = __builtin_amdgcn_wmma_f32_16x16x32_f16(false, a0.h, false, f0.h, (short)0, acc[0][0], false, false);
    acc[0][1] = __builtin_amdgcn_wmma_f32_16x16x32_f16(false, a0.h, false, f1.h, (short)0, acc[0][1], false, false);
    acc[0][2] = __builtin_amdgcn_wmma_f32_16x16x32_f16(false, a0.h, false, f2.h, (short)0, acc[0][2], false, false);
    acc[0][3] = __builtin_amdgcn_wmma_f32_16x16x32_f16(false, a0.h, false, f3.h, (short)0, acc[0][3], false, false);
    acc[1][0] = __builtin_amdgcn_wmma_f32_16x16x32_f16(false, a1.h, false, f0.h, (short)0, acc[1][0], false, false);
    acc[1][1] = __builtin_amdgcn_wmma_f32_16x16x32_f16(false, a1.h, false, f1.h, (short)0, acc[1][1], false, false);
    acc[1][2] = __builtin_amdgcn_wmma_f32_16x16x32_f16(false, a1.h, false, f2.h, (short)0, acc[1][2], false, false);
    acc[1][3] = __builtin_amdgcn_wmma_f32_16x16x32_f16(false, a1.h, false, f3.h, (short)0, acc[1][3], false, false);
  }
#pragma unroll
  for (int m = 0; m < 2; ++m) {
    int row0 = mB * 32 + m * 16 + ((lane < 16) ? 0 : 8);
#pragma unroll
    for (int n = 0; n < 4; ++n) {
      int col = nC * 64 + n * 16 + (lane & 15);
      float bv = bias ? bias[col] : 0.f;
#pragma unroll
      for (int r = 0; r < 8; ++r) {
        float vv = acc[m][n][r] + bv;
        if (GELU) vv = 0.5f * vv * (1.f + erff(vv * 0.70710678118654752f));
        size_t o = (size_t)(row0 + r) * N + col;
        if (OUT32) out32[o] = vv;
        if (OUT16) out16[o] = (_Float16)vv;
      }
    }
  }
}

// ------- shifted-window cosine attention; 1 block (64 thr) per window -------
template <int HD>
__global__ __launch_bounds__(64) void win_attn_kernel(
    const _Float16* __restrict__ qkv, const float* __restrict__ logit_scale,
    const float* __restrict__ bias16, _Float16* __restrict__ out,
    int Bn, int H, int W) {
  constexpr int NHEAD = 4, N = 16, C = NHEAD * HD;
  __shared__ float ks[NHEAD][N][HD];
  __shared__ float vs[NHEAD][N][HD];
  __shared__ int lab[N];
  int nWw = W >> 2, nWh = H >> 2;
  int win = blockIdx.x;
  int b = win / (nWh * nWw);
  int wrem = win % (nWh * nWw);
  int wh = wrem / nWw, ww = wrem % nWw;
  int t = threadIdx.x;
  int head = t >> 4, i = t & 15;
  int hs = wh * 4 + (i >> 2);          // shifted-canvas coords (mask labels)
  int wsc = ww * 4 + (i & 3);
  int h = (hs + 4) % H;                // original coords (roll by SHIFT=4)
  int w = (wsc + 4) % W;
  const _Float16* row = qkv + ((size_t)((b * H + h) * W + w)) * (3 * C);
  float q[HD], kv[HD];
  float qs = 0.f, kn = 0.f;
#pragma unroll
  for (int d = 0; d < HD; ++d) {
    float qd = (float)row[head * HD + d];
    float kd = (float)row[C + head * HD + d];
    q[d] = qd; kv[d] = kd;
    qs += qd * qd; kn += kd * kd;
  }
  float qinv = 1.f / fmaxf(sqrtf(qs), 1e-12f);
  float kinv = 1.f / fmaxf(sqrtf(kn), 1e-12f);
#pragma unroll
  for (int d = 0; d < HD; ++d) {
    q[d] *= qinv;
    ks[head][i][d] = kv[d] * kinv;
    vs[head][i][d] = (float)row[2 * C + head * HD + d];
  }
  if (head == 0) lab[i] = ((hs >= H - 4) ? 2 : 0) + ((wsc >= W - 4) ? 1 : 0);
  __syncthreads();
  float scale = expf(fminf(logit_scale[head], LOG100));
  float sc[N]; float mx = -1e30f;
  int li = lab[i];
#pragma unroll
  for (int j = 0; j < N; ++j) {
    float d = 0.f;
#pragma unroll
    for (int e = 0; e < HD; ++e) d += q[e] * ks[head][j][e];
    float val = d * scale + bias16[(head * 16 + i) * 16 + j] + ((lab[j] != li) ? -100.f : 0.f);
    sc[j] = val; mx = fmaxf(mx, val);
  }
  float sum = 0.f;
#pragma unroll
  for (int j = 0; j < N; ++j) { sc[j] = expf(sc[j] - mx); sum += sc[j]; }
  float pinv = 1.f / sum;
  _Float16* orow = out + ((size_t)((b * H + h) * W + w)) * C + head * HD;  // reverse roll+partition
#pragma unroll
  for (int d = 0; d < HD; ++d) {
    float o = 0.f;
#pragma unroll
    for (int j = 0; j < N; ++j) o += sc[j] * vs[head][j][d];
    orow[d] = (_Float16)(o * pinv);
  }
}

// ------- PatchMergingV2 gather: (B,H,W,C)f16 -> (B*H/2*W/2, 4C)f16 -------
__global__ void merge_gather_kernel(const _Float16* __restrict__ xin, _Float16* __restrict__ outm,
                                    int Bn, int H, int W, int C) {
  size_t idx = (size_t)blockIdx.x * blockDim.x + threadIdx.x;
  size_t total = (size_t)Bn * (H / 2) * (W / 2) * 4 * C;
  if (idx >= total) return;
  int C4 = 4 * C;
  int c4 = (int)(idx % C4);
  size_t rowi = idx / C4;
  int j = (int)(rowi % (W / 2));
  size_t tmp = rowi / (W / 2);
  int i = (int)(tmp % (H / 2));
  int b = (int)(tmp / (H / 2));
  int chunk = c4 / C, cc = c4 % C;
  int dh = chunk & 1;       // order: [0::2,0::2],[1::2,0::2],[0::2,1::2],[1::2,1::2]
  int dw = chunk >> 1;
  outm[idx] = xin[((size_t)((b * H + 2 * i + dh) * W) + (2 * j + dw)) * C + cc];
}

static inline unsigned cdiv(long long a, long long b) { return (unsigned)((a + b - 1) / b); }

extern "C" void kernel_launch(void* const* d_in, const int* in_sizes, int n_in,
                              void* d_out, int out_size, void* d_ws, size_t ws_size,
                              hipStream_t stream) {
  (void)in_sizes; (void)n_in; (void)out_size; (void)ws_size;
  const float* x      = (const float*)d_in[0];
  const float* conv_w = (const float*)d_in[1];
  const float* conv_b = (const float*)d_in[2];
  const float* ln0_g  = (const float*)d_in[3];
  const float* ln0_b  = (const float*)d_in[4];
  struct Blk {
    const float *qkv_w, *qkv_b, *proj_w, *proj_b, *logit_scale;
    const float *cpb_w1, *cpb_b1, *cpb_w2;
    const float *n1_g, *n1_b, *n2_g, *n2_b;
    const float *mlp_w1, *mlp_b1, *mlp_w2, *mlp_b2;
  };
  auto getBlk = [&](int base) {
    Blk k;
    k.qkv_w = (const float*)d_in[base + 0];  k.qkv_b = (const float*)d_in[base + 1];
    k.proj_w = (const float*)d_in[base + 2]; k.proj_b = (const float*)d_in[base + 3];
    k.logit_scale = (const float*)d_in[base + 4];
    k.cpb_w1 = (const float*)d_in[base + 5]; k.cpb_b1 = (const float*)d_in[base + 6];
    k.cpb_w2 = (const float*)d_in[base + 7];
    k.n1_g = (const float*)d_in[base + 8];   k.n1_b = (const float*)d_in[base + 9];
    k.n2_g = (const float*)d_in[base + 10];  k.n2_b = (const float*)d_in[base + 11];
    k.mlp_w1 = (const float*)d_in[base + 12]; k.mlp_b1 = (const float*)d_in[base + 13];
    k.mlp_w2 = (const float*)d_in[base + 14]; k.mlp_b2 = (const float*)d_in[base + 15];
    return k;
  };
  Blk blk1 = getBlk(5), blk2 = getBlk(24);
  const float* red_w1 = (const float*)d_in[21];
  const float* mg1_g  = (const float*)d_in[22];
  const float* mg1_b  = (const float*)d_in[23];
  const float* red_w2 = (const float*)d_in[40];
  const float* mg2_g  = (const float*)d_in[41];
  const float* mg2_b  = (const float*)d_in[42];

  char* base = (char*)d_ws;
  size_t cur = 0;
  auto alloc = [&](size_t bytes) -> char* {
    cur = (cur + 255) & ~(size_t)255;
    char* p = base + cur;
    cur += bytes;
    return p;
  };

  const int B = 4;
  const int H1 = 256, W1 = 256;
  const int H2 = 128, W2 = 128;
  const long long T1 = (long long)B * H1 * W1;   // 262144, C=64
  const long long T2 = (long long)B * H2 * W2;   // 65536,  C=128
  const long long T3 = T2 / 4;                   // 16384,  C=256

  // ---------------- persistent: f16 weights, bias tables, e (cross-block) ---
  auto cvt = [&](const float* src, int n) -> _Float16* {
    _Float16* dst = (_Float16*)alloc((size_t)n * 2);
    cvt_f16_kernel<<<cdiv(n, 256), 256, 0, stream>>>(src, dst, n);
    return dst;
  };
  _Float16* qkvw1  = cvt(blk1.qkv_w, 192 * 64);
  _Float16* projw1 = cvt(blk1.proj_w, 64 * 64);
  _Float16* mw11   = cvt(blk1.mlp_w1, 256 * 64);
  _Float16* mw12   = cvt(blk1.mlp_w2, 64 * 256);
  _Float16* redw1  = cvt(red_w1, 128 * 256);
  _Float16* qkvw2  = cvt(blk2.qkv_w, 384 * 128);
  _Float16* projw2 = cvt(blk2.proj_w, 128 * 128);
  _Float16* mw21   = cvt(blk2.mlp_w1, 512 * 128);
  _Float16* mw22   = cvt(blk2.mlp_w2, 128 * 512);
  _Float16* redw2  = cvt(red_w2, 256 * 512);
  float* bqkv1 = (float*)alloc(192 * 4);
  float* bqkv2 = (float*)alloc(384 * 4);
  prep_bqkv_kernel<<<1, 256, 0, stream>>>(blk1.qkv_b, bqkv1, 64);
  prep_bqkv_kernel<<<2, 256, 0, stream>>>(blk2.qkv_b, bqkv2, 128);
  float* cpb1 = (float*)alloc(1024 * 4);
  float* cpb2 = (float*)alloc(1024 * 4);
  cpb_kernel<<<1, 256, 0, stream>>>(blk1.cpb_w1, blk1.cpb_b1, blk1.cpb_w2, cpb1);
  cpb_kernel<<<1, 256, 0, stream>>>(blk2.cpb_w1, blk2.cpb_b1, blk2.cpb_w2, cpb2);
  float*    e32 = (float*)alloc((size_t)T2 * 128 * 4);   // stage-1 output (block2 input)
  _Float16* e16 = (_Float16*)alloc((size_t)T2 * 128 * 2);

  size_t mark = cur;  // scratch region reused between blocks

  // =================== stage 1: patch embed + block1 (d=64) =================
  {
    float*    p32   = (float*)alloc((size_t)T1 * 64 * 4);
    _Float16* p16   = (_Float16*)alloc((size_t)T1 * 64 * 2);
    _Float16* qkv16 = (_Float16*)alloc((size_t)T1 * 192 * 2);
    _Float16* att16 = (_Float16*)alloc((size_t)T1 * 64 * 2);
    float*    g32   = (float*)alloc((size_t)T1 * 64 * 4);
    float*    x132  = (float*)alloc((size_t)T1 * 64 * 4);
    _Float16* x116  = (_Float16*)alloc((size_t)T1 * 64 * 2);
    _Float16* h16   = (_Float16*)alloc((size_t)T1 * 256 * 2);
    float*    m32   = (float*)alloc((size_t)T1 * 64 * 4);
    _Float16* xb16  = (_Float16*)alloc((size_t)T1 * 64 * 2);
    _Float16* mg16  = (_Float16*)alloc((size_t)T2 * 256 * 2);
    float*    r32   = (float*)alloc((size_t)T2 * 128 * 4);

    patch_embed_kernel<<<cdiv(T1 * 32, 256), 256, 0, stream>>>(
        x, conv_w, conv_b, ln0_g, ln0_b, p32, p16, B, H1, W1);
    gemm_wmma_kernel<false, false, true><<<cdiv((T1 / 32) * 3, 8), 256, 0, stream>>>(
        p16, qkvw1, bqkv1, nullptr, qkv16, (int)T1, 192, 64);
    win_attn_kernel<16><<<B * (H1 / 4) * (W1 / 4), 64, 0, stream>>>(
        qkv16, blk1.logit_scale, cpb1, att16, B, H1, W1);
    gemm_wmma_kernel<false, true, false><<<cdiv(T1 / 32, 8), 256, 0, stream>>>(
        att16, projw1, blk1.proj_b, g32, nullptr, (int)T1, 64, 64);
    ln_kernel<64><<<cdiv(T1 * 32, 256), 256, 0, stream>>>(
        g32, p32, blk1.n1_g, blk1.n1_b, x132, x116, (int)T1);
    gemm_wmma_kernel<true, false, true><<<cdiv((T1 / 32) * 4, 8), 256, 0, stream>>>(
        x116, mw11, blk1.mlp_b1, nullptr, h16, (int)T1, 256, 64);
    gemm_wmma_kernel<false, true, false><<<cdiv(T1 / 32, 8), 256, 0, stream>>>(
        h16, mw12, blk1.mlp_b2, m32, nullptr, (int)T1, 64, 256);
    ln_kernel<64><<<cdiv(T1 * 32, 256), 256, 0, stream>>>(
        m32, x132, blk1.n2_g, blk1.n2_b, nullptr, xb16, (int)T1);
    merge_gather_kernel<<<cdiv(T2 * 256, 256), 256, 0, stream>>>(xb16, mg16, B, H1, W1, 64);
    gemm_wmma_kernel<false, true, false><<<cdiv((T2 / 32) * 2, 8), 256, 0, stream>>>(
        mg16, redw1, nullptr, r32, nullptr, (int)T2, 128, 256);
    ln_kernel<128><<<cdiv(T2 * 32, 256), 256, 0, stream>>>(
        r32, nullptr, mg1_g, mg1_b, e32, e16, (int)T2);
  }

  cur = mark;  // reuse scratch

  // ======================= stage 2: block2 (d=128) ===========================
  {
    _Float16* qkv16 = (_Float16*)alloc((size_t)T2 * 384 * 2);
    _Float16* att16 = (_Float16*)alloc((size_t)T2 * 128 * 2);
    float*    g32   = (float*)alloc((size_t)T2 * 128 * 4);
    float*    x232  = (float*)alloc((size_t)T2 * 128 * 4);
    _Float16* x216  = (_Float16*)alloc((size_t)T2 * 128 * 2);
    _Float16* h16   = (_Float16*)alloc((size_t)T2 * 512 * 2);
    float*    m32   = (float*)alloc((size_t)T2 * 128 * 4);
    _Float16* xb16  = (_Float16*)alloc((size_t)T2 * 128 * 2);
    _Float16* mg16  = (_Float16*)alloc((size_t)T3 * 512 * 2);
    float*    r32   = (float*)alloc((size_t)T3 * 256 * 4);

    gemm_wmma_kernel<false, false, true><<<cdiv((T2 / 32) * 6, 8), 256, 0, stream>>>(
        e16, qkvw2, bqkv2, nullptr, qkv16, (int)T2, 384, 128);
    win_attn_kernel<32><<<B * (H2 / 4) * (W2 / 4), 64, 0, stream>>>(
        qkv16, blk2.logit_scale, cpb2, att16, B, H2, W2);
    gemm_wmma_kernel<false, true, false><<<cdiv((T2 / 32) * 2, 8), 256, 0, stream>>>(
        att16, projw2, blk2.proj_b, g32, nullptr, (int)T2, 128, 128);
    ln_kernel<128><<<cdiv(T2 * 32, 256), 256, 0, stream>>>(
        g32, e32, blk2.n1_g, blk2.n1_b, x232, x216, (int)T2);
    gemm_wmma_kernel<true, false, true><<<cdiv((T2 / 32) * 8, 8), 256, 0, stream>>>(
        x216, mw21, blk2.mlp_b1, nullptr, h16, (int)T2, 512, 128);
    gemm_wmma_kernel<false, true, false><<<cdiv((T2 / 32) * 2, 8), 256, 0, stream>>>(
        h16, mw22, blk2.mlp_b2, m32, nullptr, (int)T2, 128, 512);
    ln_kernel<128><<<cdiv(T2 * 32, 256), 256, 0, stream>>>(
        m32, x232, blk2.n2_g, blk2.n2_b, nullptr, xb16, (int)T2);
    merge_gather_kernel<<<cdiv(T3 * 512, 256), 256, 0, stream>>>(xb16, mg16, B, H2, W2, 128);
    gemm_wmma_kernel<false, true, false><<<cdiv((T3 / 32) * 4, 8), 256, 0, stream>>>(
        mg16, redw2, nullptr, r32, nullptr, (int)T3, 256, 512);
    ln_kernel<256><<<cdiv(T3 * 32, 256), 256, 0, stream>>>(
        r32, nullptr, mg2_g, mg2_b, (float*)d_out, nullptr, (int)T3);
  }
}
